// FlowGuidedDeformConvInterpolation_6579889897619
// MI455X (gfx1250) — matrix operand order repaired and checked by reading, not compiled, not run on hardware
//
#include <hip/hip_runtime.h>
#include <hip/hip_bf16.h>
#include <math.h>

// ---------------------------------------------------------------------------
// FlowGuidedDeformConvInterpolation for MI455X (gfx1250, wave32, WMMA bf16)
//  - Weights pre-packed once into fragment-ready bf16 in global memory
//    (L2-resident, shared by all pixel tiles) -> A fragments via 2x
//    global_load_b128, no LDS/barriers for weights.
//  - Input patches double-buffered in LDS -> 1 barrier per K-chunk.
//  - mdcn GEMM loop is barrier-free (samples fully staged first).
// ---------------------------------------------------------------------------

#define HH 256
#define WW 256
#define NPIX (HH * WW)
#define KK 9

typedef __bf16 bf16x16 __attribute__((ext_vector_type(16)));
typedef float  f32x8   __attribute__((ext_vector_type(8)));

#if __has_builtin(__builtin_amdgcn_cvt_pk_bf16_f32)
typedef __bf16 bf16x2 __attribute__((ext_vector_type(2)));
__device__ __forceinline__ unsigned int pack2bf(float a, float b) {
    union { bf16x2 v; unsigned int u; } cv;
    cv.v = __builtin_amdgcn_cvt_pk_bf16_f32(a, b);
    return cv.u;
}
__device__ __forceinline__ unsigned short f2bf(float f) {
    return (unsigned short)(pack2bf(f, 0.0f) & 0xFFFFu);
}
#else
__device__ __forceinline__ unsigned short f2bf(float f) {
    union { float f; unsigned int u; } cv;
    cv.f = f;
    unsigned int u = cv.u;
    u += 0x7FFFu + ((u >> 16) & 1u);   // round-to-nearest-even fp32 -> bf16
    return (unsigned short)(u >> 16);
}
__device__ __forceinline__ unsigned int pack2bf(float a, float b) {
    return (unsigned int)f2bf(a) | ((unsigned int)f2bf(b) << 16);
}
#endif

// ---------------------------------------------------------------------------
// Weight pre-pack: fp32 [Cout][Ktot] -> fragment-ready bf16 at
//   wp[ ((kc*nCt + (co>>4))*32 + lane)*16 + slot ]
// in this code's contiguous k order (16-bit A 16x32): lane = (co&15) + 16*(k>>4),
//                                  slot = k&15.
// Rows co >= Cout are zero-filled (ragged Cout=216 case).
// ---------------------------------------------------------------------------
__global__ __launch_bounds__(256) void pack_wA_kernel(
    const float* __restrict__ w, unsigned short* __restrict__ wp,
    int Cout, int Ktot)
{
    const int coPad       = (Cout + 63) & ~63;
    const int nCt         = coPad >> 4;
    const int pairsPerCo  = Ktot >> 1;
    const int total       = coPad * pairsPerCo;
    int idx = blockIdx.x * 256 + threadIdx.x;
    if (idx >= total) return;

    int co = idx / pairsPerCo;
    int kp = idx - co * pairsPerCo;
    int k0 = kp << 1;
    int cgc = (co < Cout) ? co : (Cout - 1);
    float2 wv = ((const float2*)(w + (long)cgc * Ktot))[kp];  // even idx: 8B ok
    float w0 = (co < Cout) ? wv.x : 0.0f;
    float w1 = (co < Cout) ? wv.y : 0.0f;

    int kc     = k0 >> 5;
    int kl     = k0 & 31;
    int ct     = co >> 4;
    int lane_w = (co & 15) + ((kl >> 4) << 4);
    int slot   = kl & 15;       // even
    *(unsigned int*)(wp + (((long)(kc * nCt + ct) * 32 + lane_w) * 16 + slot)) =
        pack2bf(w0, w1);
}

// ---------------------------------------------------------------------------
// Implicit-GEMM conv3x3 (stride1 pad1, NCHW), bf16 WMMA, fp32 accumulate.
// WG: 256 threads (8 waves). Tile: 64 out-channels x 64 pixels.
// Wave w: co-tile (w&3), pixel-tiles {2*(w>>2), 2*(w>>2)+1}.
// A fragments: direct global_load_b128 from pre-packed weights.
// B fragments: double-buffered LDS, fragment-ready layout
//   (k,n): lane = n + 16*(k>>4), slot = k&15.
// ---------------------------------------------------------------------------
__device__ __forceinline__ void conv_load_pairs(
    const float* __restrict__ in, int kcBase, int row, int col0, int tid,
    float2 v[4])
{
#pragma unroll
    for (int j = 0; j < 4; ++j) {
        int pidx = j * 256 + tid;
        int kp   = pidx >> 6;           // 0..15
        int pi   = pidx & 63;
        int k0   = kp << 1;
        float vv[2];
#pragma unroll
        for (int h = 0; h < 2; ++h) {
            int kg = kcBase + k0 + h;
            int ci = kg / 9;
            int kk = kg - ci * 9;
            int iy = row + kk / 3 - 1;
            int ix = col0 + pi + (kk % 3) - 1;
            bool valid = ((unsigned)iy < (unsigned)HH) &&
                         ((unsigned)ix < (unsigned)WW);
            int iyc = min(max(iy, 0), HH - 1);
            int ixc = min(max(ix, 0), WW - 1);
            float t = in[(long)ci * NPIX + iyc * WW + ixc];
            vv[h] = valid ? t : 0.0f;   // select, no exec branch
        }
        v[j].x = vv[0];
        v[j].y = vv[1];
    }
}

__device__ __forceinline__ void conv_store_pairs(
    unsigned short* __restrict__ buf, int tid, const float2 v[4])
{
#pragma unroll
    for (int j = 0; j < 4; ++j) {
        int pidx   = j * 256 + tid;
        int kp     = pidx >> 6;
        int pi     = pidx & 63;
        int k0     = kp << 1;
        int pt     = pi >> 4;
        int lane_w = (pi & 15) + ((k0 >> 4) << 4);
        int slot   = k0 & 15;           // even
        *(unsigned int*)(buf + ((pt * 32 + lane_w) * 16 + slot)) =
            pack2bf(v[j].x, v[j].y);
    }
}

__global__ __launch_bounds__(256) void conv3x3_wmma_kernel(
    const float* __restrict__ in, const unsigned short* __restrict__ wpk,
    const float* __restrict__ bias, float* __restrict__ out,
    int Cin, int Cout, int nCt, int doLrelu)
{
    __shared__ alignas(32) unsigned short lds_inB[2][4 * 32 * 16];  // 2 x 4 KB

    const int tid    = threadIdx.x;
    const int lane   = tid & 31;
    const int wave   = tid >> 5;
    const int pxBase = blockIdx.x * 64;
    const int coBase = blockIdx.y * 64;
    const int row    = pxBase >> 8;     // all 64 px in one image row
    const int col0   = pxBase & 255;

    const int Ktot    = Cin * 9;        // multiple of 32 here
    const int nChunks = Ktot >> 5;

    const int coT  = wave & 3;
    const int pxT0 = (wave >> 2) << 1;
    const int ctG  = (coBase >> 4) + coT;   // global co-tile for A fetch

    f32x8 acc0 = {};
    f32x8 acc1 = {};

    // prologue: stage chunk 0
    {
        float2 v[4];
        conv_load_pairs(in, 0, row, col0, tid, v);
        conv_store_pairs(lds_inB[0], tid, v);
    }
    __syncthreads();

    for (int kc = 0; kc < nChunks; ++kc) {
        const int cur = kc & 1;

        // prefetch next chunk's input pairs (global loads overlap WMMA)
        float2 v[4];
        const bool more = (kc + 1) < nChunks;
        if (more) conv_load_pairs(in, (kc + 1) << 5, row, col0, tid, v);

        // A: 32B fragment straight from pre-packed global weights (L2 hit)
        bf16x16 a = *(const bf16x16*)(
            wpk + (((long)(kc * nCt + ctG) * 32 + lane) * 16));
        bf16x16 b0 = *(const bf16x16*)(lds_inB[cur] + (pxT0 * 32 + lane) * 16);
        bf16x16 b1 = *(const bf16x16*)(lds_inB[cur] + ((pxT0 + 1) * 32 + lane) * 16);

        acc0 = __builtin_amdgcn_wmma_f32_16x16x32_bf16(
            false, a, false, b0, (short)0, acc0, false, false);
        acc1 = __builtin_amdgcn_wmma_f32_16x16x32_bf16(
            false, a, false, b1, (short)0, acc1, false, false);

        if (more) conv_store_pairs(lds_inB[cur ^ 1], tid, v);
        __syncthreads();
    }

    // epilogue: bias (+ leaky relu); C/D layout (ISA 7.12.2)
    const int lg   = lane >> 4;
    const int ncol = lane & 15;
#pragma unroll
    for (int r = 0; r < 8; ++r) {
        int m  = r + 8 * lg;
        int co = coBase + coT * 16 + m;
        if (co < Cout) {
            float bv = bias[co];
            float v0 = acc0[r] + bv;
            float v1 = acc1[r] + bv;
            if (doLrelu) {
                v0 = (v0 >= 0.0f) ? v0 : 0.1f * v0;
                v1 = (v1 >= 0.0f) ? v1 : 0.1f * v1;
            }
            out[(long)co * NPIX + pxBase + pxT0 * 16 + ncol]       = v0;
            out[(long)co * NPIX + pxBase + (pxT0 + 1) * 16 + ncol] = v1;
        }
    }
}

// ---------------------------------------------------------------------------
// Modulated deformable conv: fused offset transform (10*tanh + flow[::-1]),
// sigmoid mask, bilinear sampling -> bf16 LDS (B-fragment-ready), then the
// 64x576 per-pixel contraction as a barrier-free WMMA GEMM (A from pre-packed
// global weights, nCt == 4).
// WG: 256 threads (8 waves). Tile: 32 pixels. Wave w: co-tile (w&3),
// px-tile (w>>2).
// ---------------------------------------------------------------------------
__global__ __launch_bounds__(256) void mdcn_wmma_kernel(
    const float* __restrict__ feat,            // [64][H][W]
    const float* __restrict__ raw,             // conv3 out [216][H][W]
    const float* __restrict__ flow,            // [2][H][W]
    const unsigned short* __restrict__ wpk,    // pre-packed dcn weights
    const float* __restrict__ db,              // [64]
    float* __restrict__ out)                   // [64][H][W]
{
    __shared__ alignas(32) unsigned short lds_SB[18 * 2 * 32 * 16];  // 36 KB

    const int tid    = threadIdx.x;
    const int lane   = tid & 31;
    const int wave   = tid >> 5;
    const int pxBase = blockIdx.x * 32;
    const int row    = pxBase >> 8;
    const int col0   = pxBase & 255;

    // ---------------- Phase 1: bilinear sampling into LDS ----------------
#pragma unroll
    for (int j = 0; j < 9; ++j) {
        int id = j * 256 + tid;
        int gk = id >> 5;               // 0..71
        int pi = id & 31;
        int g  = gk / KK;
        int k  = gk - g * KK;
        int p  = pxBase + pi;

        float oy_r = raw[(long)(g * 18 + 2 * k) * NPIX + p];
        float ox_r = raw[(long)(g * 18 + 2 * k + 1) * NPIX + p];
        float mr   = raw[(long)(144 + g * KK + k) * NPIX + p];
        float fl0  = flow[p];
        float fl1  = flow[NPIX + p];
        // tile(flow[:, ::-1]): even offset channel += flow1, odd += flow0
        float oy  = 10.0f * tanhf(oy_r) + fl1;
        float ox  = 10.0f * tanhf(ox_r) + fl0;
        float msk = 1.0f / (1.0f + expf(-mr));

        float py = (float)row + (float)(k / 3 - 1) + oy;
        float px = (float)(col0 + pi) + (float)(k % 3 - 1) + ox;
        float y0f = floorf(py), x0f = floorf(px);
        float ly = py - y0f, lx = px - x0f;
        int y0 = (int)y0f, x0 = (int)x0f;
        int y1 = y0 + 1,   x1 = x0 + 1;

        float vy0 = ((unsigned)y0 < (unsigned)HH) ? 1.0f : 0.0f;
        float vy1 = ((unsigned)y1 < (unsigned)HH) ? 1.0f : 0.0f;
        float vx0 = ((unsigned)x0 < (unsigned)WW) ? 1.0f : 0.0f;
        float vx1 = ((unsigned)x1 < (unsigned)WW) ? 1.0f : 0.0f;
        int y0c = min(max(y0, 0), HH - 1);
        int y1c = min(max(y1, 0), HH - 1);
        int x0c = min(max(x0, 0), WW - 1);
        int x1c = min(max(x1, 0), WW - 1);

        float w00 = (1.0f - ly) * (1.0f - lx) * vy0 * vx0 * msk;
        float w01 = (1.0f - ly) * lx          * vy0 * vx1 * msk;
        float w10 = ly          * (1.0f - lx) * vy1 * vx0 * msk;
        float w11 = ly          * lx          * vy1 * vx1 * msk;
        int i00 = y0c * WW + x0c, i01 = y0c * WW + x1c;
        int i10 = y1c * WW + x0c, i11 = y1c * WW + x1c;

        int pt = pi >> 4;
        int nn = pi & 15;
#pragma unroll
        for (int c = 0; c < 8; ++c) {
            int ci = g * 8 + c;
            const float* fch = feat + (long)ci * NPIX;
            float s = w00 * fch[i00] + w01 * fch[i01] +
                      w10 * fch[i10] + w11 * fch[i11];
            int sIdx = ci * KK + k;           // matches dw layout co*576 + sIdx
            int kcs  = sIdx >> 5;
            int kl   = sIdx & 31;
            int lw   = nn + ((kl >> 4) << 4);
            lds_SB[((kcs * 2 + pt) * 32 + lw) * 16 + (kl & 15)] = f2bf(s);
        }
    }
    __syncthreads();

    // ---------------- Phase 2: barrier-free GEMM 64 x 576 x 32 ------------
    const int coT = wave & 3;
    const int pxT = wave >> 2;
    f32x8 acc = {};

#pragma unroll 3
    for (int kc = 0; kc < 18; ++kc) {
        bf16x16 a = *(const bf16x16*)(
            wpk + (((long)(kc * 4 + coT) * 32 + lane) * 16));
        bf16x16 b = *(const bf16x16*)(
            lds_SB + ((kc * 2 + pxT) * 32 + lane) * 16);
        acc = __builtin_amdgcn_wmma_f32_16x16x32_bf16(
            false, a, false, b, (short)0, acc, false, false);
    }

    const int lg   = lane >> 4;
    const int ncol = lane & 15;
#pragma unroll
    for (int r = 0; r < 8; ++r) {
        int m  = r + 8 * lg;
        int co = coT * 16 + m;
        out[(long)co * NPIX + pxBase + pxT * 16 + ncol] = acc[r] + db[co];
    }
}

// ---------------------------------------------------------------------------
// Host launcher
// ---------------------------------------------------------------------------
extern "C" void kernel_launch(void* const* d_in, const int* in_sizes, int n_in,
                              void* d_out, int out_size, void* d_ws, size_t ws_size,
                              hipStream_t stream) {
    (void)in_sizes; (void)n_in; (void)out_size; (void)ws_size;

    const float* feat_prev  = (const float*)d_in[0];
    const float* feat_next  = (const float*)d_in[1];
    const float* extra_prev = (const float*)d_in[2];
    const float* extra_next = (const float*)d_in[3];
    const float* flow_prev  = (const float*)d_in[4];
    const float* flow_next  = (const float*)d_in[5];
    const float* dcn1_w = (const float*)d_in[22];
    const float* dcn1_b = (const float*)d_in[23];
    const float* dcn2_w = (const float*)d_in[24];
    const float* dcn2_b = (const float*)d_in[25];
    const float* fus_w  = (const float*)d_in[26];
    const float* fus_b  = (const float*)d_in[27];

    float* ws   = (float*)d_ws;
    float* t0   = ws;                                  // 64  ch, 16 MB
    float* t1   = t0 + (size_t)64 * NPIX;              // 64  ch, 16 MB
    float* t216 = t1 + (size_t)64 * NPIX;              // 216 ch, 54 MB
    float* fpn  = t216 + (size_t)216 * NPIX;           // fp || fn, 32 MB
    float* fp   = fpn;
    float* fn   = fpn + (size_t)64 * NPIX;
    // packed-weight scratch (ushort), 32B-aligned offsets
    unsigned short* wpkA = (unsigned short*)(fpn + (size_t)128 * NPIX); // 288KB max
    unsigned short* wpkB = wpkA + (size_t)256 * 1152;                   // 72KB

    dim3 blk(256);
    auto pack = [&](const float* w, unsigned short* wp, int Cout, int Ktot) {
        int coPad = (Cout + 63) & ~63;
        int total = coPad * (Ktot >> 1);
        hipLaunchKernelGGL(pack_wA_kernel, dim3((total + 255) / 256), blk, 0,
                           stream, w, wp, Cout, Ktot);
    };
    auto conv = [&](const float* in, int Cin, const float* w, const float* b,
                    float* out, int Cout, int lrelu) {
        pack(w, wpkA, Cout, Cin * 9);
        int nCt = ((Cout + 63) & ~63) >> 4;
        dim3 grid(NPIX / 64, (unsigned)((Cout + 63) / 64));
        hipLaunchKernelGGL(conv3x3_wmma_kernel, grid, blk, 0, stream,
                           in, wpkA, b, out, Cin, Cout, nCt, lrelu);
    };

    for (int br = 0; br < 2; ++br) {
        const float* feat  = br == 0 ? feat_prev  : feat_next;
        const float* extra = br == 0 ? extra_prev : extra_next;
        const float* flow  = br == 0 ? flow_prev  : flow_next;
        const int base = br == 0 ? 6 : 14;
        const float* w0 = (const float*)d_in[base + 0];
        const float* b0 = (const float*)d_in[base + 1];
        const float* w1 = (const float*)d_in[base + 2];
        const float* b1 = (const float*)d_in[base + 3];
        const float* w2 = (const float*)d_in[base + 4];
        const float* b2 = (const float*)d_in[base + 5];
        const float* w3 = (const float*)d_in[base + 6];
        const float* b3 = (const float*)d_in[base + 7];
        const float* dwp = br == 0 ? dcn1_w : dcn2_w;
        const float* dbp = br == 0 ? dcn1_b : dcn2_b;
        float* dst = br == 0 ? fp : fn;

        conv(extra, 128, w0, b0, t0, 64, 1);
        conv(t0,     64, w1, b1, t1, 64, 1);
        conv(t1,     64, w2, b2, t0, 64, 1);
        conv(t0,     64, w3, b3, t216, 216, 0);

        pack(dwp, wpkB, 64, 576);       // dcn weights -> fragment layout
        hipLaunchKernelGGL(mdcn_wmma_kernel, dim3(NPIX / 32), blk, 0, stream,
                           feat, t216, flow, wpkB, dbp, dst);
    }

    // fusion conv on concat(fp, fn) -> d_out (fp||fn contiguous => free concat)
    conv(fpn, 128, fus_w, fus_b, (float*)d_out, 64, 0);
}
